// PoinTrPCTransformer_72705206386854
// MI455X (gfx1250) — matrix operand order repaired and checked by reading, not compiled, not run on hardware
//
#include <hip/hip_runtime.h>
#include <hip/hip_bf16.h>
#include <math.h>

typedef long long ll;
typedef __attribute__((ext_vector_type(16))) _Float16 v16h;
typedef __attribute__((ext_vector_type(8)))  float    v8f;

// ============================ device helpers ============================

__device__ __forceinline__ float act_apply(float v, int act) {
  switch (act) {
    case 1: return v > 0.f ? v : 0.2f * v;                      // leaky relu 0.2
    case 2: {                                                    // gelu (tanh approx, jax default)
      float u = 0.7978845608028654f * (v + 0.044715f * v * v * v);
      return 0.5f * v * (1.0f + tanhf(u));
    }
    case 3: return v > 0.f ? v : 0.f;                            // relu
    default: return v;
  }
}

// ============================ WMMA GEMM =================================
// D[bo,hi][m,n] = act( alpha * sum_k A[m,k]*B[k,n] + bias )
// Block: 128 threads = 4 waves; block tile 32(M) x 128(N); wave wv owns a
// 32x32 output (2x2 of 16x16) -> 4 v_wmma per K-step per wave.
// Per K-step: cooperatively stage A (32x32) and B^T (128x32) as f16 in LDS
// (uniform fast path for interior tiles), then read WMMA fragments as
// contiguous LDS runs:
//   A frag (f16 16x32): lane L (M=L&15, half=L>>4), elem e -> K=(e<8?0:16)+half*8+(e&7)
//   B frag (f16 32x16): lane L (N=L&15, half=L>>4), elem e -> K=(e<8?0:16)+half*8+(e&7)
//   C/D   (f32 16x16):  elem r -> M = r + 8*half, N = lane&15

struct GemmArgs {
  const float* A; const float* B; const float* bias; float* D;
  int M, N, K;
  int lda, ldb, ldd;
  ll sAo, sBo, sDo, sAi, sBi, sDi;
  int H, transB, biasMode, act;   // biasMode: 0 none, 1 per-col(N), 2 per-row(M)
  float alpha;
};

#define LDSP 40   // padded row pitch in halves (80B: 16B-aligned, bank-spread)

union Frag  { v16h v; _Float16 e[16]; };
union FAcc  { v8f  v; float    e[8];  };

__global__ __launch_bounds__(128) void wmma_gemm_k(GemmArgs g) {
  __shared__ _Float16 Alds[32 * LDSP];
  __shared__ _Float16 Blds[128 * LDSP];

  const int tid  = threadIdx.x;
  const int lane = tid & 31;
  const int wv   = tid >> 5;
  const int m0   = blockIdx.x * 32;
  const int n0b  = blockIdx.y * 128;
  const int nw   = n0b + wv * 32;       // wave's 32-col window
  const int z    = blockIdx.z;
  const int bo   = z / g.H, hi = z % g.H;
  const float* __restrict__ A = g.A + (ll)bo * g.sAo + (ll)hi * g.sAi;
  const float* __restrict__ B = g.B + (ll)bo * g.sBo + (ll)hi * g.sBi;
  float* __restrict__ D = g.D + (ll)bo * g.sDo + (ll)hi * g.sDi;

  const bool waveActive = (nw < g.N);
  const int half = lane >> 4;
  const int l16  = lane & 15;

  Frag fa0, fa1, fb0, fb1;
  FAcc a00, a01, a10, a11;
#pragma unroll
  for (int r = 0; r < 8; ++r) { a00.e[r] = 0.f; a01.e[r] = 0.f; a10.e[r] = 0.f; a11.e[r] = 0.f; }

  for (int k0 = 0; k0 < g.K; k0 += 32) {
    const bool kFull = (k0 + 32 <= g.K);
    // ---- stage A tile (32 rows x 32 k): 8 elems/thread ----
    if (kFull && (m0 + 32 <= g.M)) {
#pragma unroll
      for (int i = tid; i < 32 * 32; i += 128) {
        int r = i >> 5, kk = i & 31;
        Alds[r * LDSP + kk] = (_Float16)A[(ll)(m0 + r) * g.lda + (k0 + kk)];
      }
    } else {
#pragma unroll
      for (int i = tid; i < 32 * 32; i += 128) {
        int r = i >> 5, kk = i & 31;
        int gr = m0 + r, gk = k0 + kk;
        float v = (gr < g.M && gk < g.K) ? A[(ll)gr * g.lda + gk] : 0.f;
        Alds[r * LDSP + kk] = (_Float16)v;
      }
    }
    // ---- stage B tile transposed (128 cols x 32 k): 32 elems/thread ----
    if (g.transB) {
      // global B is [N,K]: lanes walk k for coalescing
      if (kFull && (n0b + 128 <= g.N)) {
#pragma unroll
        for (int i = tid; i < 128 * 32; i += 128) {
          int kk = i & 31, n = i >> 5;
          Blds[n * LDSP + kk] = (_Float16)B[(ll)(n0b + n) * g.ldb + (k0 + kk)];
        }
      } else {
#pragma unroll
        for (int i = tid; i < 128 * 32; i += 128) {
          int kk = i & 31, n = i >> 5;
          int gn = n0b + n, gk = k0 + kk;
          float v = (gn < g.N && gk < g.K) ? B[(ll)gn * g.ldb + gk] : 0.f;
          Blds[n * LDSP + kk] = (_Float16)v;
        }
      }
    } else {
      // global B is [K,N]: lanes walk n for coalescing
      if (kFull && (n0b + 128 <= g.N)) {
#pragma unroll
        for (int i = tid; i < 128 * 32; i += 128) {
          int n = i & 127, kk = i >> 7;
          Blds[n * LDSP + kk] = (_Float16)B[(ll)(k0 + kk) * g.ldb + (n0b + n)];
        }
      } else {
#pragma unroll
        for (int i = tid; i < 128 * 32; i += 128) {
          int n = i & 127, kk = i >> 7;
          int gn = n0b + n, gk = k0 + kk;
          float v = (gn < g.N && gk < g.K) ? B[(ll)gk * g.ldb + gn] : 0.f;
          Blds[n * LDSP + kk] = (_Float16)v;
        }
      }
    }
    __syncthreads();

    if (waveActive) {
      const _Float16* ap0 = &Alds[(l16)      * LDSP + half * 8];
      const _Float16* ap1 = &Alds[(16 + l16) * LDSP + half * 8];
      const _Float16* bp0 = &Blds[(wv * 32 + l16)      * LDSP + half * 8];
      const _Float16* bp1 = &Blds[(wv * 32 + 16 + l16) * LDSP + half * 8];
#pragma unroll
      for (int e = 0; e < 8; ++e) {
        fa0.e[e] = ap0[e]; fa0.e[8 + e] = ap0[16 + e];
        fa1.e[e] = ap1[e]; fa1.e[8 + e] = ap1[16 + e];
      }
#pragma unroll
      for (int e = 0; e < 8; ++e) { fb0.e[e] = bp0[e]; fb0.e[8 + e] = bp0[16 + e]; fb1.e[e] = bp1[e]; fb1.e[8 + e] = bp1[16 + e]; }
      a00.v = __builtin_amdgcn_wmma_f32_16x16x32_f16(false, fa0.v, false, fb0.v, (short)0, a00.v, false, false);
      a01.v = __builtin_amdgcn_wmma_f32_16x16x32_f16(false, fa0.v, false, fb1.v, (short)0, a01.v, false, false);
      a10.v = __builtin_amdgcn_wmma_f32_16x16x32_f16(false, fa1.v, false, fb0.v, (short)0, a10.v, false, false);
      a11.v = __builtin_amdgcn_wmma_f32_16x16x32_f16(false, fa1.v, false, fb1.v, (short)0, a11.v, false, false);
    }
    __syncthreads();
  }

  if (waveActive) {
#pragma unroll
    for (int tm = 0; tm < 2; ++tm) {
#pragma unroll
      for (int tn = 0; tn < 2; ++tn) {
        const FAcc& ac = tm ? (tn ? a11 : a10) : (tn ? a01 : a00);
        int col = nw + tn * 16 + l16;
#pragma unroll
        for (int r = 0; r < 8; ++r) {
          int row = m0 + tm * 16 + half * 8 + r;
          if (row < g.M && col < g.N) {
            float v = ac.e[r] * g.alpha;
            if (g.bias) v += (g.biasMode == 2) ? g.bias[row] : g.bias[col];
            D[(ll)row * g.ldd + col] = act_apply(v, g.act);
          }
        }
      }
    }
  }
}

// ============================ elementwise ===============================

__global__ void add_k(const float* a, const float* b, float* o, ll n) {
  for (ll i = blockIdx.x * (ll)blockDim.x + threadIdx.x; i < n;
       i += (ll)gridDim.x * blockDim.x)
    o[i] = a[i] + b[i];
}

__global__ void copy_rows_k(float* dst, ll ldd, const float* src, ll lds, ll rows, ll cols) {
  ll n = rows * cols;
  for (ll i = blockIdx.x * (ll)blockDim.x + threadIdx.x; i < n;
       i += (ll)gridDim.x * blockDim.x) {
    ll r = i / cols, c = i % cols;
    dst[r * ldd + c] = src[r * lds + c];
  }
}

// out[b,c,r] = in[b,r,c]  (in: [B,R,C] -> out: [B,C,R])
__global__ void transpose_k(const float* in, float* out, int Bn, int R, int C) {
  ll n = (ll)Bn * R * C;
  for (ll i = blockIdx.x * (ll)blockDim.x + threadIdx.x; i < n;
       i += (ll)gridDim.x * blockDim.x) {
    int r = (int)(i % R);
    ll  t = i / R;
    int c = (int)(t % C);
    int b = (int)(t / C);
    out[i] = in[((ll)b * R + r) * C + c];
  }
}

// ============================ norms =====================================

__global__ __launch_bounds__(32) void layernorm_k(const float* x, const float* g,
                                                  const float* b, float* o, int C) {
  ll row = blockIdx.x;
  const float* xr = x + row * (ll)C;
  float s = 0.f, ss = 0.f;
  for (int c = threadIdx.x; c < C; c += 32) { float v = xr[c]; s += v; ss += v * v; }
  for (int d = 16; d; d >>= 1) { s += __shfl_xor(s, d, 32); ss += __shfl_xor(ss, d, 32); }
  float m = s / C;
  float var = ss / C - m * m;
  float r = rsqrtf(var + 1e-5f);
  float* orow = o + row * (ll)C;
  for (int c = threadIdx.x; c < C; c += 32)
    orow[c] = (xr[c] - m) * r * g[c] + b[c];
}

__global__ __launch_bounds__(32) void softmax_k(float* x, int L) {
  ll row = blockIdx.x;
  float* p = x + row * (ll)L;
  float mx = -3.4e38f;
  for (int c = threadIdx.x; c < L; c += 32) mx = fmaxf(mx, p[c]);
  for (int d = 16; d; d >>= 1) mx = fmaxf(mx, __shfl_xor(mx, d, 32));
  float s = 0.f;
  for (int c = threadIdx.x; c < L; c += 32) { float e = __expf(p[c] - mx); p[c] = e; s += e; }
  for (int d = 16; d; d >>= 1) s += __shfl_xor(s, d, 32);
  float inv = 1.f / s;
  for (int c = threadIdx.x; c < L; c += 32) p[c] *= inv;
}

// batch-norm over (B, L) per channel; x: [B,C,L]; stats[c]={mean, rstd}
__global__ __launch_bounds__(256) void bnorm_stats_k(const float* x, float* stats,
                                                     int B, int C, int L) {
  __shared__ float sh[256], sh2[256];
  int c = blockIdx.x;
  ll cnt = (ll)B * L;
  float s = 0.f, ss = 0.f;
  for (ll i = threadIdx.x; i < cnt; i += 256) {
    int b = (int)(i / L), l = (int)(i % L);
    float v = x[((ll)b * C + c) * L + l];
    s += v; ss += v * v;
  }
  sh[threadIdx.x] = s; sh2[threadIdx.x] = ss;
  __syncthreads();
  for (int o = 128; o; o >>= 1) {
    if (threadIdx.x < o) { sh[threadIdx.x] += sh[threadIdx.x + o]; sh2[threadIdx.x] += sh2[threadIdx.x + o]; }
    __syncthreads();
  }
  if (threadIdx.x == 0) {
    float m = sh[0] / cnt;
    float var = sh2[0] / cnt - m * m;
    stats[2 * c] = m;
    stats[2 * c + 1] = rsqrtf(var + 1e-5f);
  }
}

__global__ void bnorm_apply_lrelu_k(const float* x, const float* stats, const float* g,
                                    const float* b, float* o, int C, int L, ll n) {
  for (ll i = blockIdx.x * (ll)blockDim.x + threadIdx.x; i < n;
       i += (ll)gridDim.x * blockDim.x) {
    int c = (int)((i / L) % C);
    float v = (x[i] - stats[2 * c]) * stats[2 * c + 1] * g[c] + b[c];
    o[i] = v > 0.f ? v : 0.2f * v;
  }
}

// group-norm stats: y [B,O,Mn,k], 4 groups; grid = B*4
__global__ __launch_bounds__(256) void gnorm_stats_k(const float* y, float* stats,
                                                     int O, int Mn, int kk) {
  __shared__ float sh[256], sh2[256];
  int b = blockIdx.x >> 2, grp = blockIdx.x & 3;
  int Cg = O / 4;
  ll cnt = (ll)Cg * Mn * kk;
  const float* base = y + ((ll)b * O + (ll)grp * Cg) * Mn * kk;
  float s = 0.f, ss = 0.f;
  for (ll i = threadIdx.x; i < cnt; i += 256) { float v = base[i]; s += v; ss += v * v; }
  sh[threadIdx.x] = s; sh2[threadIdx.x] = ss;
  __syncthreads();
  for (int o = 128; o; o >>= 1) {
    if (threadIdx.x < o) { sh[threadIdx.x] += sh[threadIdx.x + o]; sh2[threadIdx.x] += sh2[threadIdx.x + o]; }
    __syncthreads();
  }
  if (threadIdx.x == 0) {
    float m = sh[0] / cnt;
    float var = sh2[0] / cnt - m * m;
    stats[2 * blockIdx.x] = m;
    stats[2 * blockIdx.x + 1] = rsqrtf(var + 1e-5f);
  }
}

// apply gnorm scale/shift, lrelu, then max over k -> out [B,O,Mn]
__global__ void gnorm_lrelu_max_k(const float* y, const float* stats, const float* g,
                                  const float* b, float* out, int O, int Mn, int kk, ll n) {
  for (ll i = blockIdx.x * (ll)blockDim.x + threadIdx.x; i < n;
       i += (ll)gridDim.x * blockDim.x) {
    int m = (int)(i % Mn);
    ll  t = i / Mn;
    int o = (int)(t % O);
    int bb = (int)(t / O);
    int grp = o / (O / 4);
    float mean = stats[2 * (bb * 4 + grp)];
    float rs   = stats[2 * (bb * 4 + grp) + 1];
    const float* p = y + (((ll)bb * O + o) * Mn + m) * kk;
    float best = -3.4e38f;
    for (int j = 0; j < kk; ++j) {
      float v = (p[j] - mean) * rs * g[o] + b[o];
      v = v > 0.f ? v : 0.2f * v;
      best = fmaxf(best, v);
    }
    out[((ll)bb * O + o) * Mn + m] = best;
  }
}

// ============================ point ops =================================

// cq [B,3,Mq], ck [B,3,Nk] -> idx [B,Mq,KK] of KK nearest (ascending distance)
template <int KK>
__global__ void knn_k(const float* cq, const float* ck, int* idx, int B, int Mq, int Nk) {
  ll t = blockIdx.x * (ll)blockDim.x + threadIdx.x;
  if (t >= (ll)B * Mq) return;
  int b = (int)(t / Mq), m = (int)(t % Mq);
  float qx = cq[((ll)b * 3 + 0) * Mq + m];
  float qy = cq[((ll)b * 3 + 1) * Mq + m];
  float qz = cq[((ll)b * 3 + 2) * Mq + m];
  float kd[KK]; int ki[KK];
#pragma unroll
  for (int j = 0; j < KK; ++j) { kd[j] = 3.4e38f; ki[j] = 0; }
  const float* kx = ck + (ll)b * 3 * Nk;
  for (int n = 0; n < Nk; ++n) {
    float dx = kx[n] - qx, dy = kx[Nk + n] - qy, dz = kx[2 * Nk + n] - qz;
    float d = dx * dx + dy * dy + dz * dz;
    if (d < kd[KK - 1]) {
      int j = KK - 1;
      while (j > 0 && kd[j - 1] > d) { kd[j] = kd[j - 1]; ki[j] = ki[j - 1]; --j; }
      kd[j] = d; ki[j] = n;
    }
  }
#pragma unroll
  for (int j = 0; j < KK; ++j) idx[((ll)b * Mq + m) * KK + j] = ki[j];
}

// farthest point sampling: coor [B,3,N] -> out [B,np]; one block per batch
__global__ __launch_bounds__(256) void fps_k(const float* coor, int* out, int N, int np) {
  __shared__ float dists[4096];
  __shared__ float rv[256];
  __shared__ int   ri[256];
  int b = blockIdx.x, t = threadIdx.x;
  const float* cx = coor + (ll)b * 3 * N;
  for (int i = t; i < N; i += 256) dists[i] = 1e10f;
  __syncthreads();
  int far = 0;
  for (int s = 0; s < np; ++s) {
    if (t == 0) out[(ll)b * np + s] = far;
    float fx = cx[far], fy = cx[N + far], fz = cx[2 * N + far];
    float bm = -1.f; int bi = 0;
    for (int i = t; i < N; i += 256) {
      float dx = cx[i] - fx, dy = cx[N + i] - fy, dz = cx[2 * N + i] - fz;
      float d = dx * dx + dy * dy + dz * dz;
      float nd = fminf(dists[i], d);
      dists[i] = nd;
      if (nd > bm) { bm = nd; bi = i; }
    }
    rv[t] = bm; ri[t] = bi;
    __syncthreads();
    for (int o = 128; o; o >>= 1) {
      if (t < o) {
        if (rv[t + o] > rv[t] || (rv[t + o] == rv[t] && ri[t + o] < ri[t])) {
          rv[t] = rv[t + o]; ri[t] = ri[t + o];
        }
      }
      __syncthreads();
    }
    far = ri[0];
    __syncthreads();
  }
}

// o[b,c,m] = x[b,c, idx[b,m]]
__global__ void gather_ch_k(const float* x, const int* idx, float* o,
                            int B, int C, int Nin, int Mout, ll n) {
  for (ll i = blockIdx.x * (ll)blockDim.x + threadIdx.x; i < n;
       i += (ll)gridDim.x * blockDim.x) {
    int m = (int)(i % Mout);
    ll  t = i / Mout;
    int c = (int)(t % C);
    int b = (int)(t / C);
    o[i] = x[((ll)b * C + c) * Nin + idx[(ll)b * Mout + m]];
  }
}

// feat [B, 2C, M, k]: cc<C: xk[b,cc,idx]-xq[b,cc,m]; else xq[b,cc-C,m]
__global__ void graph_feat_k(const float* xk, const float* xq, const int* idx, float* feat,
                             int B, int C, int Nk, int M, int kk, ll n) {
  for (ll i = blockIdx.x * (ll)blockDim.x + threadIdx.x; i < n;
       i += (ll)gridDim.x * blockDim.x) {
    int j = (int)(i % kk);
    ll  r = i / kk;
    int m = (int)(r % M);
    r /= M;
    int cc = (int)(r % (2 * C));
    int b  = (int)(r / (2 * C));
    int cs = cc < C ? cc : cc - C;
    float ctr = xq[((ll)b * C + cs) * M + m];
    float v;
    if (cc < C) {
      int id = idx[((ll)b * M + m) * kk + j];
      v = xk[((ll)b * C + cc) * Nk + id] - ctr;
    } else v = ctr;
    feat[i] = v;
  }
}

// token-space knn feature rows: out[(b*Nq+m)*KA+j, 2C]
__global__ void knnfeat_build_k(const float* src, const float* ctr, const int* idx, float* out,
                                int B, int Nq, int Ns, int C, int KA, ll n) {
  for (ll i = blockIdx.x * (ll)blockDim.x + threadIdx.x; i < n;
       i += (ll)gridDim.x * blockDim.x) {
    int c2 = (int)(i % (2 * C));
    ll  r = i / (2 * C);
    int j = (int)(r % KA);
    r /= KA;
    int m = (int)(r % Nq);
    int b = (int)(r / Nq);
    int cs = c2 < C ? c2 : c2 - C;
    float cv = ctr[((ll)b * Nq + m) * C + cs];
    float v;
    if (c2 < C) {
      int id = idx[((ll)b * Nq + m) * KA + j];
      v = src[((ll)b * Ns + id) * C + c2] - cv;
    } else v = cv;
    out[i] = v;
  }
}

// o[r,c] = max_j x[(r*KA+j)*C + c]
__global__ void rowgroup_max_k(const float* x, float* o, ll R, int KA, int C, ll n) {
  for (ll i = blockIdx.x * (ll)blockDim.x + threadIdx.x; i < n;
       i += (ll)gridDim.x * blockDim.x) {
    ll r = i / C;
    int c = (int)(i % C);
    float best = -3.4e38f;
    for (int j = 0; j < KA; ++j) best = fmaxf(best, x[(r * KA + j) * (ll)C + c]);
    o[i] = best;
  }
}

// o[r] = max over L of x[r*L + j]
__global__ void maxlast_k(const float* x, float* o, ll rows, int L) {
  for (ll i = blockIdx.x * (ll)blockDim.x + threadIdx.x; i < rows;
       i += (ll)gridDim.x * blockDim.x) {
    const float* p = x + i * (ll)L;
    float m = p[0];
    for (int j = 1; j < L; ++j) m = fmaxf(m, p[j]);
    o[i] = m;
  }
}

// qin [B,1027,NQ]: ch<1024 -> gf[b,ch]; else coarse[b,m,ch-1024]
__global__ void qin_build_k(const float* gf, const float* coarse, float* qin,
                            int B, int NQv, ll n) {
  for (ll i = blockIdx.x * (ll)blockDim.x + threadIdx.x; i < n;
       i += (ll)gridDim.x * blockDim.x) {
    int m = (int)(i % NQv);
    ll  t = i / NQv;
    int ch = (int)(t % 1027);
    int b  = (int)(t / 1027);
    qin[i] = (ch < 1024) ? gf[(ll)b * 1024 + ch]
                         : coarse[((ll)b * NQv + m) * 3 + (ch - 1024)];
  }
}

// ============================ host side =================================

static inline unsigned ew_grid(ll n) {
  ll g = (n + 255) / 256;
  if (g < 1) g = 1;
  if (g > 1048576) g = 1048576;
  return (unsigned)g;
}

static void gemm(hipStream_t s, const float* A, const float* B, const float* bias, float* D,
                 int M, int N, int K, int lda, int ldb, int ldd,
                 ll sAo, ll sBo, ll sDo, ll sAi, ll sBi, ll sDi,
                 int Bo, int H, int transB, float alpha, int act, int biasMode) {
  GemmArgs g{A, B, bias, D, M, N, K, lda, ldb, ldd,
             sAo, sBo, sDo, sAi, sBi, sDi, H, transB, biasMode, act, alpha};
  dim3 grid((unsigned)((M + 31) / 32), (unsigned)((N + 127) / 128), (unsigned)(Bo * H));
  wmma_gemm_k<<<grid, dim3(128), 0, s>>>(g);
}

// token linear: X [R, Cin] @ W [Cin, Cout] (+bias per-col)
static void ling(hipStream_t s, const float* X, const float* W, const float* bias,
                 float* D, ll R, int Cin, int Cout, int act) {
  gemm(s, X, W, bias, D, (int)R, Cout, Cin, Cin, Cout, Cout,
       0, 0, 0, 0, 0, 0, 1, 1, 0, 1.f, act, bias ? 1 : 0);
}

// conv1: W [O,Cin] @ X[b][Cin,L] (+bias per-row)
static void convg(hipStream_t s, const float* W, const float* X, const float* bias,
                  float* D, int O, int Cin, int L, int Bo, int act) {
  gemm(s, W, X, bias, D, O, L, Cin, Cin, L, L,
       0, (ll)Cin * L, (ll)O * L, 0, 0, 0, Bo, 1, 0, 1.f, act, bias ? 2 : 0);
}

extern "C" void kernel_launch(void* const* d_in, const int* in_sizes, int n_in,
                              void* d_out, int out_size, void* d_ws, size_t ws_size,
                              hipStream_t stream) {
  (void)in_sizes; (void)n_in; (void)out_size; (void)ws_size;
  auto P = [&](int i) -> const float* { return (const float*)d_in[i]; };

  // ----- pytree leaf indices (jax sorted-key flatten order) -----
  // 0: incomplete_point_cloud
  // decoder: geom_ca (27 leaves) then 7x ca (19 leaves) -> [1, 161)
  // encoder: geom_sa (15) then 5x sa (11)               -> [161, 231)
  // grouper [231,245) input_proj [245,251) pos_embed [251,257) qgen [257,273)
  const int DEC0 = 1, ENC0 = 161, GRP = 231, IPJ = 245, PEM = 251, QGN = 257;

  // ----- workspace arena (floats) -----
  float* Wb = (float*)d_ws;
  ll off = 0;
  auto alloc = [&](ll n) { float* p = Wb + off; off += (n + 63) & ~63ll; return p; };

  float* BIGA = alloc(16777216);   // graph feats / knn feat rows / qgen temps
  float* BIGB = alloc(16777216);   // edge-conv pre-norm / knn feat gemm out
  float* coorA = alloc(98304);     // [8,3,4096]
  float* f0    = alloc(262144);    // [8,8,4096]  (also reused [8,128,256])
  float* f1    = alloc(1048576);   // [8,32,4096]
  int*   idx16 = (int*)alloc(524288);
  int*   idxfp = (int*)alloc(8192);
  float* coorB = alloc(24576);     // [8,3,1024]
  float* fqA   = alloc(262144);    // [8,32,1024]
  float* f2    = alloc(524288);    // [8,64,1024]
  float* f3    = alloc(524288);
  float* coorC = alloc(6144);      // [8,3,256]
  float* fqB   = alloc(131072);    // [8,64,256]
  float* f4    = alloc(262144);    // [8,128,256]
  float* stats = alloc(2048);
  float* pos   = alloc(786432);    // [8,384,256]
  float* post  = alloc(786432);    // [8,256,384]
  float* x     = alloc(786432);
  float* xin   = alloc(786432);
  float* nx    = alloc(786432);
  float* t384  = alloc(786432);
  float* qkvb  = alloc(2359296);   // [8,256,1152] (dec: [8,224,1152])
  float* scor  = alloc(3145728);   // [8,6,256,256] / [8,6,224,256]
  float* obuf  = alloc(786432);
  float* x1    = alloc(786432);
  float* kf    = alloc(786432);
  float* cat2  = alloc(1572864);   // [2048,768]
  float* hid   = alloc(1572864);
  int*   idx8  = (int*)alloc(16384);
  float* gf    = alloc(8192);      // [8,1024]
  float* c1b   = alloc(8192);
  float* coarse= alloc(5440);      // [8,224,3]
  float* qin   = alloc(1840384);   // [8,1027,224]
  float* qf    = alloc(688128);    // [8,384,224]
  float* q     = alloc(688128);    // [8,224,384]
  float* qcoor = alloc(5440);      // [8,3,224]
  float* nqb   = alloc(688128);    // LN(q)
  float* nvb   = alloc(786432);    // LN(v)
  float* qtmp  = alloc(688128);
  float* kvb   = alloc(1572864);   // [8,256,768]
  float* qpro  = alloc(688128);

  const int B = 8, NPTS = 4096, C = 384, H = 6, NT = 256, NQv = 224;
  const float SC = 0.125f; // 64^-0.5

  // =================== DGCNN grouper ===================
  transpose_k<<<ew_grid((ll)B*NPTS*3), 256, 0, stream>>>(P(0), coorA, B, NPTS, 3);
  convg(stream, P(GRP+1), coorA, P(GRP+0), f0, 8, 3, NPTS, B, 0);     // input_trans
  // l1
  knn_k<16><<<ew_grid((ll)B*NPTS), 256, 0, stream>>>(coorA, coorA, idx16, B, NPTS, NPTS);
  graph_feat_k<<<ew_grid((ll)B*16*NPTS*16), 256, 0, stream>>>(f0, f0, idx16, BIGA,
                                                              B, 8, NPTS, NPTS, 16, (ll)B*16*NPTS*16);
  convg(stream, P(GRP+4), BIGA, nullptr, BIGB, 32, 16, NPTS*16, B, 0);
  gnorm_stats_k<<<B*4, 256, 0, stream>>>(BIGB, stats, 32, NPTS, 16);
  gnorm_lrelu_max_k<<<ew_grid((ll)B*32*NPTS), 256, 0, stream>>>(BIGB, stats, P(GRP+3), P(GRP+2),
                                                                f1, 32, NPTS, 16, (ll)B*32*NPTS);
  // fps 4096 -> 1024
  fps_k<<<B, 256, 0, stream>>>(coorA, idxfp, NPTS, 1024);
  gather_ch_k<<<ew_grid((ll)B*3*1024), 256, 0, stream>>>(coorA, idxfp, coorB, B, 3, NPTS, 1024, (ll)B*3*1024);
  gather_ch_k<<<ew_grid((ll)B*32*1024), 256, 0, stream>>>(f1, idxfp, fqA, B, 32, NPTS, 1024, (ll)B*32*1024);
  // l2
  knn_k<16><<<ew_grid((ll)B*1024), 256, 0, stream>>>(coorB, coorA, idx16, B, 1024, NPTS);
  graph_feat_k<<<ew_grid((ll)B*64*1024*16), 256, 0, stream>>>(f1, fqA, idx16, BIGA,
                                                              B, 32, NPTS, 1024, 16, (ll)B*64*1024*16);
  convg(stream, P(GRP+7), BIGA, nullptr, BIGB, 64, 64, 1024*16, B, 0);
  gnorm_stats_k<<<B*4, 256, 0, stream>>>(BIGB, stats, 64, 1024, 16);
  gnorm_lrelu_max_k<<<ew_grid((ll)B*64*1024), 256, 0, stream>>>(BIGB, stats, P(GRP+6), P(GRP+5),
                                                                f2, 64, 1024, 16, (ll)B*64*1024);
  // l3
  knn_k<16><<<ew_grid((ll)B*1024), 256, 0, stream>>>(coorB, coorB, idx16, B, 1024, 1024);
  graph_feat_k<<<ew_grid((ll)B*128*1024*16), 256, 0, stream>>>(f2, f2, idx16, BIGA,
                                                               B, 64, 1024, 1024, 16, (ll)B*128*1024*16);
  convg(stream, P(GRP+10), BIGA, nullptr, BIGB, 64, 128, 1024*16, B, 0);
  gnorm_stats_k<<<B*4, 256, 0, stream>>>(BIGB, stats, 64, 1024, 16);
  gnorm_lrelu_max_k<<<ew_grid((ll)B*64*1024), 256, 0, stream>>>(BIGB, stats, P(GRP+9), P(GRP+8),
                                                                f3, 64, 1024, 16, (ll)B*64*1024);
  // fps 1024 -> 256
  fps_k<<<B, 256, 0, stream>>>(coorB, idxfp, 1024, 256);
  gather_ch_k<<<ew_grid((ll)B*3*256), 256, 0, stream>>>(coorB, idxfp, coorC, B, 3, 1024, 256, (ll)B*3*256);
  gather_ch_k<<<ew_grid((ll)B*64*256), 256, 0, stream>>>(f3, idxfp, fqB, B, 64, 1024, 256, (ll)B*64*256);
  // l4
  knn_k<16><<<ew_grid((ll)B*256), 256, 0, stream>>>(coorC, coorB, idx16, B, 256, 1024);
  graph_feat_k<<<ew_grid((ll)B*128*256*16), 256, 0, stream>>>(f3, fqB, idx16, BIGA,
                                                              B, 64, 1024, 256, 16, (ll)B*128*256*16);
  convg(stream, P(GRP+13), BIGA, nullptr, BIGB, 128, 128, 256*16, B, 0);
  gnorm_stats_k<<<B*4, 256, 0, stream>>>(BIGB, stats, 128, 256, 16);
  gnorm_lrelu_max_k<<<ew_grid((ll)B*128*256), 256, 0, stream>>>(BIGB, stats, P(GRP+12), P(GRP+11),
                                                                f4, 128, 256, 16, (ll)B*128*256);

  // =================== pos embed / input proj ===================
  convg(stream, P(PEM+3), coorC, P(PEM+2), f0, 128, 3, NT, B, 0);     // pe.c1 -> f0 [8,128,256]
  bnorm_stats_k<<<128, 256, 0, stream>>>(f0, stats, B, 128, NT);
  bnorm_apply_lrelu_k<<<ew_grid((ll)B*128*NT), 256, 0, stream>>>(f0, stats, P(PEM+1), P(PEM+0),
                                                                 f0, 128, NT, (ll)B*128*NT);
  convg(stream, P(PEM+5), f0, P(PEM+4), pos, C, 128, NT, B, 0);       // pe.c2
  transpose_k<<<ew_grid((ll)B*C*NT), 256, 0, stream>>>(pos, post, B, C, NT);

  convg(stream, P(IPJ+3), f4, P(IPJ+2), x1, C, 128, NT, B, 0);        // ip.c1
  bnorm_stats_k<<<C, 256, 0, stream>>>(x1, stats, B, C, NT);
  bnorm_apply_lrelu_k<<<ew_grid((ll)B*C*NT), 256, 0, stream>>>(x1, stats, P(IPJ+1), P(IPJ+0),
                                                               x1, C, NT, (ll)B*C*NT);
  convg(stream, P(IPJ+5), x1, P(IPJ+4), t384, C, C, NT, B, 0);        // ip.c2
  transpose_k<<<ew_grid((ll)B*C*NT), 256, 0, stream>>>(t384, x, B, C, NT); // x [8,256,384]

  // =================== encoder ===================
  for (int i = 0; i < 6; ++i) {
    int pb = (i == 0) ? ENC0 : ENC0 + 15 + (i - 1) * 11;
    const float *Wpb, *Wpw, *Wqkv, *Wf1b, *Wf1w, *Wf2b, *Wf2w, *Wn1b, *Wn1g, *Wn2b, *Wn2g;
    const float *Wknb = nullptr, *Wknw = nullptr, *Wmeb = nullptr, *Wmew = nullptr;
    if (i == 0) {
      Wpb = P(pb+0); Wpw = P(pb+1); Wqkv = P(pb+2);
      Wknb = P(pb+3); Wknw = P(pb+4); Wmeb = P(pb+5); Wmew = P(pb+6);
      Wf1b = P(pb+7); Wf1w = P(pb+8); Wf2b = P(pb+9); Wf2w = P(pb+10);
      Wn1b = P(pb+11); Wn1g = P(pb+12); Wn2b = P(pb+13); Wn2g = P(pb+14);
    } else {
      Wpb = P(pb+0); Wpw = P(pb+1); Wqkv = P(pb+2);
      Wf1b = P(pb+3); Wf1w = P(pb+4); Wf2b = P(pb+5); Wf2w = P(pb+6);
      Wn1b = P(pb+7); Wn1g = P(pb+8); Wn2b = P(pb+9); Wn2g = P(pb+10);
    }
    add_k<<<ew_grid((ll)B*NT*C), 256, 0, stream>>>(x, post, xin, (ll)B*NT*C);
    layernorm_k<<<B*NT, 32, 0, stream>>>(xin, Wn1g, Wn1b, nx, C);
    // self attention
    ling(stream, nx, Wqkv, nullptr, qkvb, (ll)B*NT, C, 3*C, 0);
    gemm(stream, qkvb, qkvb + C, nullptr, scor, NT, NT, 64, 1152, 1152, NT,
         (ll)NT*1152, (ll)NT*1152, (ll)H*NT*NT, 64, 64, (ll)NT*NT, B, H, 1, SC, 0, 0);
    softmax_k<<<B*H*NT, 32, 0, stream>>>(scor, NT);
    gemm(stream, scor, qkvb + 2*C, nullptr, obuf, NT, 64, NT, NT, 1152, C,
         (ll)H*NT*NT, (ll)NT*1152, (ll)NT*C, (ll)NT*NT, 64, 64, B, H, 0, 1.f, 0, 0);
    ling(stream, obuf, Wpw, Wpb, x1, (ll)B*NT, C, C, 0);
    if (i == 0) {
      knn_k<8><<<ew_grid((ll)B*NT), 256, 0, stream>>>(coorC, coorC, idx8, B, NT, NT);
      knnfeat_build_k<<<ew_grid((ll)B*NT*8*2*C), 256, 0, stream>>>(nx, nx, idx8, BIGA,
                                                                   B, NT, NT, C, 8, (ll)B*NT*8*2*C);
      ling(stream, BIGA, Wknw, Wknb, BIGB, (ll)B*NT*8, 2*C, C, 1);
      rowgroup_max_k<<<ew_grid((ll)B*NT*C), 256, 0, stream>>>(BIGB, kf, (ll)B*NT, 8, C, (ll)B*NT*C);
      copy_rows_k<<<ew_grid((ll)B*NT*C), 256, 0, stream>>>(cat2, 2*C, x1, C, (ll)B*NT, C);
      copy_rows_k<<<ew_grid((ll)B*NT*C), 256, 0, stream>>>(cat2 + C, 2*C, kf, C, (ll)B*NT, C);
      ling(stream, cat2, Wmew, Wmeb, t384, (ll)B*NT, 2*C, C, 0);
      add_k<<<ew_grid((ll)B*NT*C), 256, 0, stream>>>(xin, t384, x, (ll)B*NT*C);
    } else {
      add_k<<<ew_grid((ll)B*NT*C), 256, 0, stream>>>(xin, x1, x, (ll)B*NT*C);
    }
    // mlp
    layernorm_k<<<B*NT, 32, 0, stream>>>(x, Wn2g, Wn2b, nx, C);
    ling(stream, nx, Wf1w, Wf1b, hid, (ll)B*NT, C, 2*C, 2);
    ling(stream, hid, Wf2w, Wf2b, x1, (ll)B*NT, 2*C, C, 0);
    add_k<<<ew_grid((ll)B*NT*C), 256, 0, stream>>>(x, x1, x, (ll)B*NT*C);
  }

  // =================== query generator ===================
  transpose_k<<<ew_grid((ll)B*NT*C), 256, 0, stream>>>(x, nx, B, NT, C);  // featT [8,384,256]
  float* t1a = BIGA;
  float* t1b = BIGA + 2097152;
  convg(stream, P(QGN+9), nx, P(QGN+8), t1a, 1024, C, NT, B, 0);          // inc1
  bnorm_stats_k<<<1024, 256, 0, stream>>>(t1a, stats, B, 1024, NT);
  bnorm_apply_lrelu_k<<<ew_grid((ll)B*1024*NT), 256, 0, stream>>>(t1a, stats, P(QGN+1), P(QGN+0),
                                                                  t1a, 1024, NT, (ll)B*1024*NT);
  convg(stream, P(QGN+11), t1a, P(QGN+10), t1b, 1024, 1024, NT, B, 0);    // inc2
  maxlast_k<<<ew_grid((ll)B*1024), 256, 0, stream>>>(t1b, gf, (ll)B*1024, NT);
  ling(stream, gf, P(QGN+5), P(QGN+4), c1b, B, 1024, 1024, 3);            // coarse1 + relu
  ling(stream, c1b, P(QGN+7), P(QGN+6), coarse, B, 1024, 3*NQv, 0);       // coarse2
  qin_build_k<<<ew_grid((ll)B*1027*NQv), 256, 0, stream>>>(gf, coarse, qin, B, NQv, (ll)B*1027*NQv);
  convg(stream, P(QGN+13), qin, P(QGN+12), t1a, 1024, 1027, NQv, B, 0);   // mq1
  bnorm_stats_k<<<1024, 256, 0, stream>>>(t1a, stats, B, 1024, NQv);
  bnorm_apply_lrelu_k<<<ew_grid((ll)B*1024*NQv), 256, 0, stream>>>(t1a, stats, P(QGN+3), P(QGN+2),
                                                                   t1a, 1024, NQv, (ll)B*1024*NQv);
  convg(stream, P(QGN+15), t1a, P(QGN+14), qf, C, 1024, NQv, B, 0);       // mq2
  transpose_k<<<ew_grid((ll)B*C*NQv), 256, 0, stream>>>(qf, q, B, C, NQv);       // q [8,224,384]
  transpose_k<<<ew_grid((ll)B*NQv*3), 256, 0, stream>>>(coarse, qcoor, B, NQv, 3); // [8,3,224]

  // =================== decoder ===================
  for (int i = 0; i < 8; ++i) {
    int pb = (i == 0) ? DEC0 : DEC0 + 27 + (i - 1) * 19;
    const float *Wkv, *Wcpb, *Wcpw, *Wcq, *Wf1b, *Wf1w, *Wf2b, *Wf2w;
    const float *Wn1b, *Wn1g, *Wn2b, *Wn2g, *Wnqb, *Wnqg, *Wnvb, *Wnvg, *Wspb, *Wspw, *Wsqkv;
    const float *Wk1b = nullptr, *Wk1w = nullptr, *Wk2b = nullptr, *Wk2w = nullptr;
    const float *Wm1b = nullptr, *Wm1w = nullptr, *Wm2b = nullptr, *Wm2w = nullptr;
    if (i == 0) {
      Wkv = P(pb+0); Wcpb = P(pb+1); Wcpw = P(pb+2); Wcq = P(pb+3);
      Wk1b = P(pb+4); Wk1w = P(pb+5); Wk2b = P(pb+6); Wk2w = P(pb+7);
      Wm1b = P(pb+8); Wm1w = P(pb+9); Wm2b = P(pb+10); Wm2w = P(pb+11);
      Wf1b = P(pb+12); Wf1w = P(pb+13); Wf2b = P(pb+14); Wf2w = P(pb+15);
      Wn1b = P(pb+16); Wn1g = P(pb+17); Wn2b = P(pb+18); Wn2g = P(pb+19);
      Wnqb = P(pb+20); Wnqg = P(pb+21); Wnvb = P(pb+22); Wnvg = P(pb+23);
      Wspb = P(pb+24); Wspw = P(pb+25); Wsqkv = P(pb+26);
    } else {
      Wkv = P(pb+0); Wcpb = P(pb+1); Wcpw = P(pb+2); Wcq = P(pb+3);
      Wf1b = P(pb+4); Wf1w = P(pb+5); Wf2b = P(pb+6); Wf2w = P(pb+7);
      Wn1b = P(pb+8); Wn1g = P(pb+9); Wn2b = P(pb+10); Wn2g = P(pb+11);
      Wnqb = P(pb+12); Wnqg = P(pb+13); Wnvb = P(pb+14); Wnvg = P(pb+15);
      Wspb = P(pb+16); Wspw = P(pb+17); Wsqkv = P(pb+18);
    }
    // ---- self attention ----
    layernorm_k<<<B*NQv, 32, 0, stream>>>(q, Wn1g, Wn1b, nqb, C);
    ling(stream, nqb, Wsqkv, nullptr, qkvb, (ll)B*NQv, C, 3*C, 0);
    gemm(stream, qkvb, qkvb + C, nullptr, scor, NQv, NQv, 64, 1152, 1152, NQv,
         (ll)NQv*1152, (ll)NQv*1152, (ll)H*NQv*NQv, 64, 64, (ll)NQv*NQv, B, H, 1, SC, 0, 0);
    softmax_k<<<B*H*NQv, 32, 0, stream>>>(scor, NQv);
    gemm(stream, scor, qkvb + 2*C, nullptr, obuf, NQv, 64, NQv, NQv, 1152, C,
         (ll)H*NQv*NQv, (ll)NQv*1152, (ll)NQv*C, (ll)NQv*NQv, 64, 64, B, H, 0, 1.f, 0, 0);
    ling(stream, obuf, Wspw, Wspb, x1, (ll)B*NQv, C, C, 0);
    if (i == 0) {
      knn_k<8><<<ew_grid((ll)B*NQv), 256, 0, stream>>>(qcoor, qcoor, idx8, B, NQv, NQv);
      knnfeat_build_k<<<ew_grid((ll)B*NQv*8*2*C), 256, 0, stream>>>(nqb, nqb, idx8, BIGA,
                                                                    B, NQv, NQv, C, 8, (ll)B*NQv*8*2*C);
      ling(stream, BIGA, Wk1w, Wk1b, BIGB, (ll)B*NQv*8, 2*C, C, 1);
      rowgroup_max_k<<<ew_grid((ll)B*NQv*C), 256, 0, stream>>>(BIGB, kf, (ll)B*NQv, 8, C, (ll)B*NQv*C);
      copy_rows_k<<<ew_grid((ll)B*NQv*C), 256, 0, stream>>>(cat2, 2*C, x1, C, (ll)B*NQv, C);
      copy_rows_k<<<ew_grid((ll)B*NQv*C), 256, 0, stream>>>(cat2 + C, 2*C, kf, C, (ll)B*NQv, C);
      ling(stream, cat2, Wm1w, Wm1b, qtmp, (ll)B*NQv, 2*C, C, 0);
      add_k<<<ew_grid((ll)B*NQv*C), 256, 0, stream>>>(q, qtmp, q, (ll)B*NQv*C);
    } else {
      add_k<<<ew_grid((ll)B*NQv*C), 256, 0, stream>>>(q, x1, q, (ll)B*NQv*C);
    }
    // ---- cross attention ----
    layernorm_k<<<B*NQv, 32, 0, stream>>>(q, Wnqg, Wnqb, nqb, C);
    layernorm_k<<<B*NT, 32, 0, stream>>>(x, Wnvg, Wnvb, nvb, C);
    ling(stream, nqb, Wcq, nullptr, qpro, (ll)B*NQv, C, C, 0);
    ling(stream, nvb, Wkv, nullptr, kvb, (ll)B*NT, C, 2*C, 0);
    gemm(stream, qpro, kvb, nullptr, scor, NQv, NT, 64, C, 768, NT,
         (ll)NQv*C, (ll)NT*768, (ll)H*NQv*NT, 64, 64, (ll)NQv*NT, B, H, 1, SC, 0, 0);
    softmax_k<<<B*H*NQv, 32, 0, stream>>>(scor, NT);
    gemm(stream, scor, kvb + C, nullptr, obuf, NQv, 64, NT, NT, 768, C,
         (ll)H*NQv*NT, (ll)NT*768, (ll)NQv*C, (ll)NQv*NT, 64, 64, B, H, 0, 1.f, 0, 0);
    ling(stream, obuf, Wcpw, Wcpb, x1, (ll)B*NQv, C, C, 0);
    if (i == 0) {
      knn_k<8><<<ew_grid((ll)B*NQv), 256, 0, stream>>>(qcoor, coorC, idx8, B, NQv, NT);
      knnfeat_build_k<<<ew_grid((ll)B*NQv*8*2*C), 256, 0, stream>>>(nvb, nqb, idx8, BIGA,
                                                                    B, NQv, NT, C, 8, (ll)B*NQv*8*2*C);
      ling(stream, BIGA, Wk2w, Wk2b, BIGB, (ll)B*NQv*8, 2*C, C, 1);
      rowgroup_max_k<<<ew_grid((ll)B*NQv*C), 256, 0, stream>>>(BIGB, kf, (ll)B*NQv, 8, C, (ll)B*NQv*C);
      copy_rows_k<<<ew_grid((ll)B*NQv*C), 256, 0, stream>>>(cat2, 2*C, x1, C, (ll)B*NQv, C);
      copy_rows_k<<<ew_grid((ll)B*NQv*C), 256, 0, stream>>>(cat2 + C, 2*C, kf, C, (ll)B*NQv, C);
      ling(stream, cat2, Wm2w, Wm2b, qtmp, (ll)B*NQv, 2*C, C, 0);
      add_k<<<ew_grid((ll)B*NQv*C), 256, 0, stream>>>(q, qtmp, q, (ll)B*NQv*C);
    } else {
      add_k<<<ew_grid((ll)B*NQv*C), 256, 0, stream>>>(q, x1, q, (ll)B*NQv*C);
    }
    // ---- mlp ----
    layernorm_k<<<B*NQv, 32, 0, stream>>>(q, Wn2g, Wn2b, nqb, C);
    ling(stream, nqb, Wf1w, Wf1b, hid, (ll)B*NQv, C, 2*C, 2);
    ling(stream, hid, Wf2w, Wf2b, x1, (ll)B*NQv, 2*C, C, 0);
    add_k<<<ew_grid((ll)B*NQv*C), 256, 0, stream>>>(q, x1, q, (ll)B*NQv*C);
  }

  // =================== outputs: (q, coarse) flat ===================
  hipMemcpyAsync(d_out, q, (size_t)B * NQv * C * sizeof(float),
                 hipMemcpyDeviceToDevice, stream);
  hipMemcpyAsync((float*)d_out + (ll)B * NQv * C, coarse,
                 (size_t)B * NQv * 3 * sizeof(float), hipMemcpyDeviceToDevice, stream);
}
